// SignLLM_68994354643360
// MI455X (gfx1250) — matrix operand order repaired and checked by reading, not compiled, not run on hardware
//
#include <hip/hip_runtime.h>
#include <math.h>

// ---------------------------------------------------------------------------
// CDNA5 (gfx1250) wave32 WMMA: D(16x16 f32) += A(16x32 f16) x B(32x16 f16)
// Fragment layouts per CDNA5 ISA 7.12.2:
//   A 16x32 (MxK): lane m = lane&15; khalf = (lane&16)?8:0;
//                  half i in [0,8): k = khalf+i ; i in [8,16): k = 16+khalf+(i-8)
//   B 32x16 (KxN): lane n = lane&15; kb = (lane&16)?8:0; half i: k as in A with khalf = kb
//   C/D 16x16 f32: row = ((lane&16)?8:0)+i ; col = lane&15
// Strategy: stage fragments in LDS in *fragment order* so the MMA loop is just
// two wide DS reads + one v_wmma per wave per 32-deep K chunk; all gather math
// and f32->f16 conversion happens in a batched cooperative fill phase.
// ---------------------------------------------------------------------------
typedef __attribute__((ext_vector_type(16))) _Float16 v16h;
typedef __attribute__((ext_vector_type(8)))  float    v8f;

#define DEV __device__ __forceinline__

DEV void tile_rc(int i, int& r, int& c) {
  int lane = threadIdx.x & 31;
  r = ((lane & 16) ? 8 : 0) + i;
  c = lane & 15;
}

// Block-tiled GEMM engine. blockDim must be 256 (8 waves), MW*NW == 8.
// Computes a (MW*16) x (NW*16) tile; this wave's result returned, its
// sub-tile offsets in (wm*16, wn*16). la(m,k)/lb(k,n) are block-local sources.
template <int MW, int NW, class LA, class LB>
DEV v8f tile_gemm(int K, LA&& la, LB&& lb, int& wm, int& wn) {
  __shared__ _Float16 Af[MW * 512];   // fragment-order: [tile][lane][16 halfs]
  __shared__ _Float16 Bf[NW * 512];
  const int tid  = threadIdx.x;
  const int lane = tid & 31;
  const int wave = tid >> 5;
  wm = wave / NW;
  wn = wave - wm * NW;
  v8f acc = {};
  for (int kc = 0; kc < K; kc += 32) {
    __syncthreads();               // protect LDS reuse across chunks
    float av[MW * 2], bv[NW * 2];
#pragma unroll
    for (int j = 0; j < MW * 2; ++j) {   // batched loads (amortize loadcnt waits)
      int e = tid + j * 256;
      int i = e & 15, tl = e >> 4, tile = tl >> 5, ln = tl & 31;
      int kh = (ln & 16) ? 8 : 0;
      int k  = (i < 8) ? (kh + i) : (16 + kh + (i - 8));
      av[j] = la(tile * 16 + (ln & 15), kc + k);
    }
#pragma unroll
    for (int j = 0; j < NW * 2; ++j) {
      int e = tid + j * 256;
      int i = e & 15, tl = e >> 4, tile = tl >> 5, ln = tl & 31;
      int kb = (ln & 16) ? 8 : 0;
      bv[j] = lb(kc + ((i < 8) ? (kb + i) : (16 + kb + (i - 8))), tile * 16 + (ln & 15));
    }
#pragma unroll
    for (int j = 0; j < MW * 2; ++j) Af[tid + j * 256] = (_Float16)av[j];
#pragma unroll
    for (int j = 0; j < NW * 2; ++j) Bf[tid + j * 256] = (_Float16)bv[j];
    __syncthreads();
    v16h a = *(const v16h*)(Af + (wm * 32 + lane) * 16);  // 32B DS read
    v16h b = *(const v16h*)(Bf + (wn * 32 + lane) * 16);
    acc = __builtin_amdgcn_wmma_f32_16x16x32_f16(false, a, false, b, (short)0, acc,
                                                 false, false);
  }
  return acc;
}

DEV float block_reduce_sum256(float v) {   // requires blockDim.x == 256
  __shared__ float sh[256];
  __syncthreads();
  sh[threadIdx.x] = v;
  __syncthreads();
  for (int st = 128; st > 0; st >>= 1) {
    if ((int)threadIdx.x < st) sh[threadIdx.x] += sh[threadIdx.x + st];
    __syncthreads();
  }
  return sh[0];
}

// ---------------------------------------------------------------------------
// Utility kernels
// ---------------------------------------------------------------------------
__global__ void k_zero(float* p, int n) {
  int i = blockIdx.x * blockDim.x + threadIdx.x;
  if (i < n) p[i] = 0.f;
}

// videos (4,3,32,64,64) -> x0[(clip*4+b), c, h, w], 16 clips, frame = clip*2
__global__ void k_gather_clips(const float* __restrict__ v, float* __restrict__ x0) {
  int i = blockIdx.x * 256 + threadIdx.x;
  if (i >= 64 * 3 * 4096) return;
  int hw = i & 4095;
  int c  = (i >> 12) % 3;
  int n  = i / (3 * 4096);
  int clip = n >> 2, b = n & 3;
  x0[i] = v[((long)(b * 3 + c) * 32 + clip * 2) * 4096 + hw];
}

// ---------------------------------------------------------------------------
// Implicit-GEMM 3x3 conv (temporal kernel=3, T=1, pad=1 -> only middle tap).
// Block tile: 64 Cout x 32 pixels via tile_gemm<4,2>. k->(c,dy,dx) via LDS LUT.
// ---------------------------------------------------------------------------
__global__ void k_conv3x3(const float* __restrict__ in, const float* __restrict__ w5,
                          const float* __restrict__ bias, float* __restrict__ out,
                          int Cin, int Cout, int H, int Wsh) {
  __shared__ short       t_c[1152];
  __shared__ signed char t_r[1152], t_dy[1152], t_dx[1152];
  const int W = 1 << Wsh, HW = H << Wsh;
  const int K = Cin * 9;
  for (int k = threadIdx.x; k < K; k += 256) {
    int c = k / 9, r = k - c * 9;
    t_c[k] = (short)c; t_r[k] = (signed char)r;
    t_dy[k] = (signed char)(r / 3 - 1); t_dx[k] = (signed char)(r % 3 - 1);
  }
  int pixblk = blockIdx.x * 32;
  int obase  = blockIdx.y * 64;
  int img    = blockIdx.z;
  const float* inp = in + (long)img * Cin * HW;
  auto la = [&](int m, int k) -> float {
    if (k >= K) return 0.f;
    return w5[(((long)(obase + m) * Cin + t_c[k]) * 3 + 1) * 9 + t_r[k]];
  };
  auto lb = [&](int k, int n) -> float {
    if (k >= K) return 0.f;
    int pix = pixblk + n;
    int h = pix >> Wsh, x = pix & (W - 1);
    int hy = h + t_dy[k], xx = x + t_dx[k];
    if ((unsigned)hy >= (unsigned)H || (unsigned)xx >= (unsigned)W) return 0.f;
    return inp[((long)t_c[k] * H + hy) * W + xx];
  };
  int wm, wn;
  v8f acc = tile_gemm<4, 2>(K, la, lb, wm, wn);
  float* op = out + (long)img * Cout * HW;
#pragma unroll
  for (int i = 0; i < 8; ++i) {
    int r, c; tile_rc(i, r, c);
    int co = obase + wm * 16 + r;
    op[(long)co * HW + pixblk + wn * 16 + c] = acc[i] + bias[co];
  }
}

// BN (training stats per clip over B,H,W) -> fused scale/shift per (clip,ch)
__global__ void k_bnstats(const float* __restrict__ x, const float* __restrict__ g,
                          const float* __restrict__ bt, float* __restrict__ scsh,
                          int C, int HW) {
  int clip = blockIdx.x / C, ch = blockIdx.x - clip * C;
  int n = 4 * HW;
  float s = 0.f, s2 = 0.f;
  for (int i = threadIdx.x; i < n; i += 256) {
    int b = i / HW, p = i - b * HW;
    float v = x[((long)(clip * 4 + b) * C + ch) * HW + p];
    s += v; s2 += v * v;
  }
  float S  = block_reduce_sum256(s);
  float S2 = block_reduce_sum256(s2);
  if (threadIdx.x == 0) {
    float m   = S / n;
    float var = S2 / n - m * m;
    float sc  = g[ch] * rsqrtf(var + 1e-5f);
    scsh[2 * blockIdx.x]     = sc;
    scsh[2 * blockIdx.x + 1] = bt[ch] - m * sc;
  }
}

// fused BN + ReLU + 2x2 maxpool
__global__ void k_bn_relu_pool(const float* __restrict__ x, const float* __restrict__ scsh,
                               float* __restrict__ out, int C, int H, int W) {
  int Ho = H >> 1, Wo = W >> 1;
  long total = 64L * C * Ho * Wo;
  long i = (long)blockIdx.x * 256 + threadIdx.x;
  if (i >= total) return;
  int wo = i % Wo; long t = i / Wo;
  int ho = t % Ho; t /= Ho;
  int ch = t % C;  int nimg = (int)(t / C);
  int clip = nimg >> 2;
  float sc = scsh[2 * (clip * C + ch)], sb = scsh[2 * (clip * C + ch) + 1];
  const float* p = x + (((long)nimg * C + ch) * H + 2 * ho) * W + 2 * wo;
  float a = fmaxf(p[0] * sc + sb, 0.f);
  a = fmaxf(a, fmaxf(p[1] * sc + sb, 0.f));
  a = fmaxf(a, fmaxf(p[W] * sc + sb, 0.f));
  a = fmaxf(a, fmaxf(p[W + 1] * sc + sb, 0.f));
  out[i] = a;
}

// fused BN + ReLU (in place)
__global__ void k_bn_relu(float* __restrict__ x, const float* __restrict__ scsh,
                          int C, int HW) {
  long total = 64L * C * HW;
  long i = (long)blockIdx.x * 256 + threadIdx.x;
  if (i >= total) return;
  long t = i / HW;
  int ch = t % C; int nimg = (int)(t / C);
  int clip = nimg >> 2;
  float sc = scsh[2 * (clip * C + ch)], sb = scsh[2 * (clip * C + ch) + 1];
  x[i] = fmaxf(x[i] * sc + sb, 0.f);
}

// AdaptiveAvgPool (16,16)->(4,4): mean of each 4x4 block
__global__ void k_avgpool4(const float* __restrict__ x, float* __restrict__ out) {
  int i = blockIdx.x * 256 + threadIdx.x;  // 64*256*16
  if (i >= 64 * 256 * 16) return;
  int wo = i & 3, ho = (i >> 2) & 3, nc_ = i >> 4;
  const float* p = x + (long)nc_ * 256 + ho * 4 * 16 + wo * 4;
  float s = 0.f;
#pragma unroll
  for (int dy = 0; dy < 4; ++dy)
#pragma unroll
    for (int dx = 0; dx < 4; ++dx) s += p[dy * 16 + dx];
  out[i] = s * (1.f / 16.f);
}

// tconv: (3,1,1) kernel, T=1 pad 1 -> 1x1 conv 256->256 on (64,256,16 px)
__global__ void k_tconv(const float* __restrict__ in, const float* __restrict__ w5,
                        const float* __restrict__ bias, float* __restrict__ out) {
  int nblk  = blockIdx.x * 32;   // N = 64 imgs * 16 px = 1024
  int obase = blockIdx.y * 64;
  auto la = [&](int m, int k) -> float { return w5[((long)(obase + m) * 256 + k) * 3 + 1]; };
  auto lb = [&](int k, int n) -> float {
    int nn = nblk + n; int img = nn >> 4, p = nn & 15;
    return in[((long)img * 256 + k) * 16 + p];
  };
  int wm, wn;
  v8f acc = tile_gemm<4, 2>(256, la, lb, wm, wn);
#pragma unroll
  for (int i = 0; i < 8; ++i) {
    int r, c; tile_rc(i, r, c);
    int co = obase + wm * 16 + r;
    int nn = nblk + wn * 16 + c; int img = nn >> 4, p = nn & 15;
    out[((long)img * 256 + co) * 16 + p] = acc[i] + bias[co];
  }
}

// proj: features[b*16+clip, d] = feats[clip*4+b, :4096] . proj_w[d, :4096] + pb[d]
__global__ void k_proj(const float* __restrict__ feats, const float* __restrict__ pw,
                       const float* __restrict__ pb, float* __restrict__ features) {
  int nblk  = blockIdx.x * 128;
  int mbase = blockIdx.y * 16;
  auto la = [&](int m, int k) -> float {
    int r = mbase + m; int b = r >> 4, clip = r & 15;
    return feats[(long)(clip * 4 + b) * 4096 + k];
  };
  auto lb = [&](int k, int n) -> float { return pw[(long)(nblk + n) * 4096 + k]; };
  int wm, wn;
  v8f acc = tile_gemm<1, 8>(4096, la, lb, wm, wn);
#pragma unroll
  for (int i = 0; i < 8; ++i) {
    int r, c; tile_rc(i, r, c);
    int cc = nblk + wn * 16 + c;
    features[(long)(mbase + r) * 512 + cc] = acc[i] + pb[cc];
  }
}

// generic Y[Mr,N] = X[Mr,K] @ W[N,K]^T (+bias) (optional relu). K % 32 == 0.
__global__ void k_gemm_xwt(const float* __restrict__ X, const float* __restrict__ W,
                           const float* __restrict__ bias, float* __restrict__ Y,
                           int Mr, int N, int K, int relu) {
  int nblk  = blockIdx.x * 128;
  int mbase = blockIdx.y * 16;
  auto la = [&](int m, int k) -> float {
    int r = mbase + m; return (r < Mr) ? X[(long)r * K + k] : 0.f;
  };
  auto lb = [&](int k, int n) -> float {
    int cN = nblk + n; return (cN < N) ? W[(long)cN * K + k] : 0.f;
  };
  int wm, wn;
  v8f acc = tile_gemm<1, 8>(K, la, lb, wm, wn);
#pragma unroll
  for (int i = 0; i < 8; ++i) {
    int r, c; tile_rc(i, r, c);
    int rr = mbase + r, cc = nblk + wn * 16 + c;
    if (rr < Mr && cc < N) {
      float v = acc[i] + (bias ? bias[cc] : 0.f);
      Y[(long)rr * N + cc] = relu ? fmaxf(v, 0.f) : v;
    }
  }
}

// per-row squared L2 norm (one wave per row)
__global__ void k_rownorm(const float* __restrict__ X, float* __restrict__ out, int K) {
  int row = blockIdx.x;
  float s = 0.f;
  for (int k = threadIdx.x; k < K; k += 32) {
    float v = X[(long)row * K + k]; s += v * v;
  }
  for (int off = 16; off; off >>= 1) s += __shfl_xor(s, off);
  if (threadIdx.x == 0) out[row] = s;
}

// argmin over codes + gather code vector + optional SSE vs ref + optional index out
__global__ void k_argmin(const float* __restrict__ G, const float* __restrict__ rn,
                         const float* __restrict__ cn, const float* __restrict__ book,
                         int Ncodes, int D, float* __restrict__ outvec,
                         float* __restrict__ outidx, const float* __restrict__ ref,
                         float* __restrict__ sseAcc) {
  int row = blockIdx.x;
  __shared__ float sval[256];
  __shared__ int   sidx[256];
  float best = 3.402e38f; int bi = 0;
  for (int j = threadIdx.x; j < Ncodes; j += 256) {
    float d = rn[row] - 2.f * G[(long)row * Ncodes + j] + cn[j];
    if (d < best) { best = d; bi = j; }
  }
  sval[threadIdx.x] = best; sidx[threadIdx.x] = bi;
  __syncthreads();
  for (int st = 128; st; st >>= 1) {
    if ((int)threadIdx.x < st) {
      float ov = sval[threadIdx.x + st]; int oi = sidx[threadIdx.x + st];
      if (ov < sval[threadIdx.x] || (ov == sval[threadIdx.x] && oi < sidx[threadIdx.x])) {
        sval[threadIdx.x] = ov; sidx[threadIdx.x] = oi;
      }
    }
    __syncthreads();
  }
  int tok = sidx[0];
  if (outidx && threadIdx.x == 0) outidx[row] = (float)tok;
  float part = 0.f;
  for (int d = threadIdx.x; d < D; d += 256) {
    float q = book[(long)tok * D + d];
    outvec[(long)row * D + d] = q;
    if (ref) { float df = ref[(long)row * D + d] - q; part += df * df; }
  }
  if (sseAcc) {
    float S = block_reduce_sum256(part);
    if (threadIdx.x == 0) atomicAdd(sseAcc, S);
  }
}

// gather GRU layer-0 input rows: x0g[b*15+t] = quant[b*16+t]
__global__ void k_gru_x0(const float* __restrict__ quant, float* __restrict__ x0g) {
  int i = blockIdx.x * 256 + threadIdx.x;  // 60*512
  if (i >= 60 * 512) return;
  int d = i & 511, r = i >> 9;
  int b = r / 15, t = r - b * 15;
  x0g[i] = quant[(long)(b * 16 + t) * 512 + d];
}

// one GRU step: gh = h @ whh^T + bhh (12 x tile_gemm<1,8> passes), then
// elementwise PyTorch GRU update. Single workgroup (latency chain), 8 waves.
__global__ void k_gru_step(const float* __restrict__ gi, const float* __restrict__ whh,
                           const float* __restrict__ bhh, float* __restrict__ h,
                           float* __restrict__ y, int t) {
  __shared__ float gh[4 * 1536];
  for (int pass = 0; pass < 12; ++pass) {
    int nblk = pass * 128;
    auto la = [&](int m, int k) -> float { return (m < 4) ? h[m * 512 + k] : 0.f; };
    auto lb = [&](int k, int n) -> float { return whh[(long)(nblk + n) * 512 + k]; };
    int wm, wn;
    v8f acc = tile_gemm<1, 8>(512, la, lb, wm, wn);
#pragma unroll
    for (int i = 0; i < 8; ++i) {
      int r, c; tile_rc(i, r, c);
      int cc = nblk + wn * 16 + c;
      if (r < 4) gh[r * 1536 + cc] = acc[i] + bhh[cc];
    }
  }
  __syncthreads();
  for (int e = threadIdx.x; e < 4 * 512; e += 256) {
    int b = e >> 9, j = e & 511;
    const float* gib = gi + (long)(b * 15 + t) * 1536;
    float ir = gib[j], iz = gib[512 + j], in_ = gib[1024 + j];
    float hr = gh[b * 1536 + j], hz = gh[b * 1536 + 512 + j], hn = gh[b * 1536 + 1024 + j];
    float r = 1.f / (1.f + __expf(-(ir + hr)));
    float z = 1.f / (1.f + __expf(-(iz + hz)));
    float n = tanhf(in_ + r * hn);
    float hv = (1.f - z) * n + z * h[e];
    h[e] = hv;
    y[(long)(b * 15 + t) * 512 + j] = hv;
  }
}

// context loss: SSE(y[b*15+t] - features[b*16+t+1])
__global__ void k_ctxloss(const float* __restrict__ y, const float* __restrict__ f,
                          float* __restrict__ acc) {
  float s = 0.f;
  for (int i = blockIdx.x * 256 + threadIdx.x; i < 60 * 512; i += gridDim.x * 256) {
    int d = i & 511, r = i >> 9;
    int b = r / 15, t = r - b * 15;
    float df = y[i] - f[(long)(b * 16 + t + 1) * 512 + d];
    s += df * df;
  }
  float S = block_reduce_sum256(s);
  if (threadIdx.x == 0) atomicAdd(acc, S);
}

__global__ void k_sqdiff(const float* __restrict__ a, const float* __restrict__ b,
                         int n, float* __restrict__ acc) {
  float s = 0.f;
  for (int i = blockIdx.x * 256 + threadIdx.x; i < n; i += gridDim.x * 256) {
    float d = a[i] - b[i]; s += d * d;
  }
  float S = block_reduce_sum256(s);
  if (threadIdx.x == 0) atomicAdd(acc, S);
}

// wsrc[b*8+w] = mean(quant[b*16+2w], quant[b*16+2w+1])
__global__ void k_wsrc(const float* __restrict__ quant, float* __restrict__ wsrc) {
  int i = blockIdx.x * 256 + threadIdx.x;  // 32*512
  if (i >= 32 * 512) return;
  int d = i & 511, r = i >> 9;
  int b = r >> 3, wi = r & 7;
  wsrc[i] = 0.5f * (quant[(long)(b * 16 + 2 * wi) * 512 + d] +
                    quant[(long)(b * 16 + 2 * wi + 1) * 512 + d]);
}

// ctx[b] = mean over 8 words of wemb
__global__ void k_ctxmean(const float* __restrict__ wemb, float* __restrict__ ctxv) {
  int i = blockIdx.x * 256 + threadIdx.x;  // 4*512
  if (i >= 2048) return;
  int d = i & 511, b = i >> 9;
  float s = 0.f;
#pragma unroll
  for (int w = 0; w < 8; ++w) s += wemb[(long)(b * 8 + w) * 512 + d];
  ctxv[i] = s * 0.125f;
}

// cross entropy (first token per row, summed over batch)
__global__ void k_ce(const float* __restrict__ lg_, const int* __restrict__ tok,
                     float* __restrict__ acc) {
  const float* lg = lg_ + (long)blockIdx.x * 3000;
  __shared__ float sh[256];
  float m = -3.402e38f;
  for (int j = threadIdx.x; j < 3000; j += 256) m = fmaxf(m, lg[j]);
  sh[threadIdx.x] = m; __syncthreads();
  for (int st = 128; st; st >>= 1) {
    if ((int)threadIdx.x < st) sh[threadIdx.x] = fmaxf(sh[threadIdx.x], sh[threadIdx.x + st]);
    __syncthreads();
  }
  float MX = sh[0]; __syncthreads();
  float s = 0.f;
  for (int j = threadIdx.x; j < 3000; j += 256) s += __expf(lg[j] - MX);
  float SE = block_reduce_sum256(s);
  if (threadIdx.x == 0) {
    float lse = MX + logf(SE);
    atomicAdd(acc, -(lg[tok[blockIdx.x * 10]] - lse));
  }
}

// total = 1.25*mse + 0.1*ctx + 0.1*align + translation  (means from SSE accumulators)
__global__ void k_final(const float* __restrict__ a, float* __restrict__ out) {
  if (threadIdx.x == 0 && blockIdx.x == 0) {
    float mse = a[0] / (64.f * 512.f);
    float ctx = a[1] / (60.f * 512.f);
    float ali = a[2] / (32.f * 512.f);
    out[32] = 1.25f * mse + 0.1f * ctx + 0.1f * ali + a[3];
  }
}

// ---------------------------------------------------------------------------
extern "C" void kernel_launch(void* const* d_in, const int* in_sizes, int n_in,
                              void* d_out, int out_size, void* d_ws, size_t ws_size,
                              hipStream_t stream) {
  (void)in_sizes; (void)n_in; (void)out_size; (void)ws_size;
  const float* videos = (const float*)d_in[0];
  const float* c1w = (const float*)d_in[1];  const float* c1b = (const float*)d_in[2];
  const float* g1  = (const float*)d_in[3];  const float* be1 = (const float*)d_in[4];
  const float* c2w = (const float*)d_in[5];  const float* c2b = (const float*)d_in[6];
  const float* g2  = (const float*)d_in[7];  const float* be2 = (const float*)d_in[8];
  const float* c3w = (const float*)d_in[9];  const float* c3b = (const float*)d_in[10];
  const float* g3  = (const float*)d_in[11]; const float* be3 = (const float*)d_in[12];
  const float* tw  = (const float*)d_in[13]; const float* tb  = (const float*)d_in[14];
  const float* g4  = (const float*)d_in[15]; const float* be4 = (const float*)d_in[16];
  const float* pw  = (const float*)d_in[17]; const float* pb  = (const float*)d_in[18];
  const float* cbk = (const float*)d_in[19]; const float* wcb = (const float*)d_in[20];
  const float* aw  = (const float*)d_in[21]; const float* ab  = (const float*)d_in[22];
  const float* wih0= (const float*)d_in[23]; const float* whh0= (const float*)d_in[24];
  const float* bih0= (const float*)d_in[25]; const float* bhh0= (const float*)d_in[26];
  const float* wih1= (const float*)d_in[27]; const float* whh1= (const float*)d_in[28];
  const float* bih1= (const float*)d_in[29]; const float* bhh1= (const float*)d_in[30];
  const float* d1w = (const float*)d_in[31]; const float* d1b = (const float*)d_in[32];
  const float* d2w = (const float*)d_in[33]; const float* d2b = (const float*)d_in[34];
  const float* d3w = (const float*)d_in[35]; const float* d3b = (const float*)d_in[36];
  const int*   toks= (const int*)d_in[37];
  float* out = (float*)d_out;

  float* ws = (float*)d_ws;
  size_t off = 0;
  auto alloc = [&](size_t n) { float* p = ws + off; off += (n + 63) & ~(size_t)63; return p; };
  float* ACC  = alloc(8);
  float* X0   = alloc(64L * 3 * 4096);        // gathered clips
  float* P2   = alloc(64L * 128 * 256);       // pooled conv2
  float* G4   = alloc(64L * 256 * 16);        // avgpool out
  float* A4   = alloc(64L * 256 * 16);        // tconv out -> feats
  float* SCSH = alloc(16L * 256 * 2);         // BN scale/shift (reused per layer)
  float* FEAT = alloc(64L * 512);
  float* CN   = alloc(1024);
  float* RN   = alloc(64);
  float* GMAT = alloc(64L * 1024);
  float* QUANT= alloc(64L * 512);
  float* X0G  = alloc(60L * 512);
  float* GI   = alloc(60L * 1536);
  float* Hb   = alloc(4L * 512);
  float* Y0   = alloc(60L * 512);
  float* Y1   = alloc(60L * 512);
  float* WSRC = alloc(32L * 512);
  float* WN   = alloc(256);
  float* WEMB = alloc(32L * 512);
  float* AE   = alloc(32L * 512);
  float* CTXV = alloc(4L * 512);
  float* D1   = alloc(4L * 512);
  float* D2   = alloc(4L * 1024);
  float* LG   = alloc(4L * 3008);
  float* BIGA = alloc(64L * 64 * 4096);       // largest pre-BN activation (conv1)
  float* BIGB = alloc(64L * 64 * 1024);       // pooled conv1

  k_zero<<<1, 64, 0, stream>>>(ACC, 8);
  k_gather_clips<<<(64 * 3 * 4096 + 255) / 256, 256, 0, stream>>>(videos, X0);

  // conv1 (K=27) -> BN stats -> BN+ReLU+pool
  k_conv3x3<<<dim3(128, 1, 64), 256, 0, stream>>>(X0, c1w, c1b, BIGA, 3, 64, 64, 6);
  k_bnstats<<<16 * 64, 256, 0, stream>>>(BIGA, g1, be1, SCSH, 64, 4096);
  k_bn_relu_pool<<<(64 * 64 * 1024 + 255) / 256, 256, 0, stream>>>(BIGA, SCSH, BIGB, 64, 64, 64);

  // conv2 (K=576)
  k_conv3x3<<<dim3(32, 2, 64), 256, 0, stream>>>(BIGB, c2w, c2b, BIGA, 64, 128, 32, 5);
  k_bnstats<<<16 * 128, 256, 0, stream>>>(BIGA, g2, be2, SCSH, 128, 1024);
  k_bn_relu_pool<<<(64 * 128 * 256 + 255) / 256, 256, 0, stream>>>(BIGA, SCSH, P2, 128, 32, 32);

  // conv3 (K=1152), no pool
  k_conv3x3<<<dim3(8, 4, 64), 256, 0, stream>>>(P2, c3w, c3b, BIGA, 128, 256, 16, 4);
  k_bnstats<<<16 * 256, 256, 0, stream>>>(BIGA, g3, be3, SCSH, 256, 256);
  k_bn_relu<<<(64 * 256 * 256 + 255) / 256, 256, 0, stream>>>(BIGA, SCSH, 256, 256);
  k_avgpool4<<<(64 * 256 * 16 + 255) / 256, 256, 0, stream>>>(BIGA, G4);

  // tconv (1x1) -> BN -> feats
  k_tconv<<<dim3(32, 4), 256, 0, stream>>>(G4, tw, tb, A4);
  k_bnstats<<<16 * 256, 256, 0, stream>>>(A4, g4, be4, SCSH, 256, 16);
  k_bn_relu<<<(64 * 256 * 16 + 255) / 256, 256, 0, stream>>>(A4, SCSH, 256, 16);

  // projection to D=512
  k_proj<<<dim3(4, 4), 256, 0, stream>>>(A4, pw, pb, FEAT);

  // VQ: dot GEMM + argmin + gather + commitment/codebook SSE
  k_rownorm<<<1024, 32, 0, stream>>>(cbk, CN, 512);
  k_rownorm<<<64, 32, 0, stream>>>(FEAT, RN, 512);
  k_gemm_xwt<<<dim3(8, 4), 256, 0, stream>>>(FEAT, cbk, nullptr, GMAT, 64, 1024, 512, 0);
  k_argmin<<<64, 256, 0, stream>>>(GMAT, RN, CN, cbk, 1024, 512, QUANT, nullptr, FEAT, ACC + 0);

  // GRU layer 0: input-gate GEMM for all steps at once, then sequential steps
  k_gru_x0<<<120, 256, 0, stream>>>(QUANT, X0G);
  k_gemm_xwt<<<dim3(12, 4), 256, 0, stream>>>(X0G, wih0, bih0, GI, 60, 1536, 512, 0);
  k_zero<<<8, 256, 0, stream>>>(Hb, 2048);
  for (int t = 0; t < 15; ++t)
    k_gru_step<<<1, 256, 0, stream>>>(GI, whh0, bhh0, Hb, Y0, t);
  // GRU layer 1
  k_gemm_xwt<<<dim3(12, 4), 256, 0, stream>>>(Y0, wih1, bih1, GI, 60, 1536, 512, 0);
  k_zero<<<8, 256, 0, stream>>>(Hb, 2048);
  for (int t = 0; t < 15; ++t)
    k_gru_step<<<1, 256, 0, stream>>>(GI, whh1, bhh1, Hb, Y1, t);
  k_ctxloss<<<30, 256, 0, stream>>>(Y1, FEAT, ACC + 1);

  // word-level CRA: pair means -> nearest word code (indices -> d_out[0..31])
  k_wsrc<<<64, 256, 0, stream>>>(QUANT, WSRC);
  k_rownorm<<<256, 32, 0, stream>>>(wcb, WN, 512);
  k_rownorm<<<32, 32, 0, stream>>>(WSRC, RN, 512);
  k_gemm_xwt<<<dim3(2, 2), 256, 0, stream>>>(WSRC, wcb, nullptr, GMAT, 32, 256, 512, 0);
  k_argmin<<<32, 256, 0, stream>>>(GMAT, RN, WN, wcb, 256, 512, WEMB, out, nullptr, nullptr);

  // alignment loss
  k_gemm_xwt<<<dim3(4, 2), 256, 0, stream>>>(WEMB, aw, ab, AE, 32, 512, 512, 0);
  k_sqdiff<<<16, 256, 0, stream>>>(AE, WEMB, 32 * 512, ACC + 2);

  // translation MLP + CE
  k_ctxmean<<<8, 256, 0, stream>>>(WEMB, CTXV);
  k_gemm_xwt<<<dim3(4, 1), 256, 0, stream>>>(CTXV, d1w, d1b, D1, 4, 512, 512, 1);
  k_gemm_xwt<<<dim3(8, 1), 256, 0, stream>>>(D1, d2w, d2b, D2, 4, 1024, 512, 1);
  k_gemm_xwt<<<dim3(24, 1), 256, 0, stream>>>(D2, d3w, d3b, LG, 4, 3000, 1024, 0);
  k_ce<<<4, 256, 0, stream>>>(LG, toks, ACC + 3);

  k_final<<<1, 32, 0, stream>>>(ACC, out);
}
